// GatedESN_49340584296810
// MI455X (gfx1250) — matrix operand order repaired and checked by reading, not compiled, not run on hardware
//
#include <hip/hip_runtime.h>
#include <hip/hip_bf16.h>

// GatedESN on MI455X (gfx1250).
// B=8 T=2048 I=128 R=1024 H=256, leaky=1 so x_t = tanh(uin_t + x W^T).
// Latency-bound serial recurrences -> LDS-resident weights (staged via TDM
// tensor_load_to_lds) + wave32 WMMA bf16.

typedef __attribute__((ext_vector_type(16))) __bf16 v16bf;
typedef __attribute__((ext_vector_type(8)))  float  v8f;
typedef __attribute__((ext_vector_type(4)))  unsigned int uint4v;
typedef __attribute__((ext_vector_type(4)))  unsigned int su4;
typedef __attribute__((ext_vector_type(8)))  unsigned int su8;

union BF16x16 { v16bf bf; uint4v q[2]; };

// ---- fragment loaders -------------------------------------------------------
// A 16x32 bf16 tile (ISA 7.12.2): lanes 0-15 rows M=lane, K groups {0..7,16..23};
// lanes 16-31 rows M=lane-16, K groups {8..15,24..31}. Two b128 loads per lane.
__device__ inline v16bf load_a_frag(const __hip_bfloat16* base, int lda, int lane) {
  int m = lane & 15, hl = lane >> 4;
  BF16x16 u;
  u.q[0] = *(const uint4v*)(base + (size_t)m * lda + hl * 8);
  u.q[1] = *(const uint4v*)(base + (size_t)m * lda + 16 + hl * 8);
  return u.bf;
}

// B fragment from pre-packed buffer: 16 contiguous bf16 per lane (32B).
__device__ inline v16bf load_b_frag(const __hip_bfloat16* lanep) {
  BF16x16 u;
  const uint4v* p = (const uint4v*)lanep;
  u.q[0] = p[0];
  u.q[1] = p[1];
  return u.bf;
}

__device__ inline v8f wmma_bf16(v16bf a, v16bf b, v8f c) {
  return __builtin_amdgcn_wmma_f32_16x16x32_bf16(false, a, false, b, (short)0, c,
                                                 false, false);
}

// ---- Tensor Data Mover: flat 1-D copy global -> LDS (n8 elements of 8 bytes)
// D# per CDNA5 ISA ch.8: group0 {count=1, lds_addr, global_addr, type=2},
// group1 {data_size=3(8B), tensor_dim0=tile_dim0=n8, dim1=1, stride0=n8}.
// Issue once per workgroup (TDM ignores EXEC); tracked by TENSORcnt.
__device__ inline void tdm_copy_to_lds(const void* gsrc, unsigned lds_byte_off,
                                       unsigned n8) {
  unsigned long long ga = (unsigned long long)gsrc;
  su4 g0;
  g0[0] = 1u;                                   // count=1 (one valid descriptor)
  g0[1] = lds_byte_off;                         // lds_addr (bytes)
  g0[2] = (unsigned)ga;                         // global_addr[31:0]
  g0[3] = ((unsigned)(ga >> 32) & 0x01FFFFFFu)  // global_addr[56:32]
          | (2u << 30);                         // type=2 ("image")
  su8 g1;
  g1[0] = 3u << 16;                 // workgroup_mask=0, data_size=3 (8 bytes)
  g1[1] = (n8 & 0xFFFFu) << 16;     // tensor_dim0[15:0] @ bits 63:48
  g1[2] = (n8 >> 16) | (1u << 16);  // tensor_dim0[31:16] | tensor_dim1=1
  g1[3] = (n8 & 0xFFFFu) << 16;     // tile_dim0 @ bits 127:112
  g1[4] = 1u;                       // tile_dim1=1, tile_dim2=0
  g1[5] = n8;                       // tensor_dim0_stride[31:0]
  g1[6] = 0u;                       // stride hi / dim1_stride lo
  g1[7] = 0u;
  asm volatile("tensor_load_to_lds %0, %1" :: "s"(g0), "s"(g1) : "memory");
}

// ---- inter-workgroup step barrier (co-resident blocks) ----------------------
__device__ inline void grid_sync(unsigned int* bar, unsigned int target) {
  __threadfence();
  __syncthreads();
  if (threadIdx.x == 0) {
    atomicAdd(bar, 1u);
    while (atomicAdd(bar, 0u) < target) { __builtin_amdgcn_s_sleep(2); }
  }
  __syncthreads();
  __threadfence();
  // NOP when not dispatched as a cluster (ClusterID==0); aligns WGs when clustered.
  __builtin_amdgcn_s_cluster_barrier();
}

// ---- packers ---------------------------------------------------------------
__global__ void f32_to_bf16_k(const float* __restrict__ src,
                              __hip_bfloat16* __restrict__ dst, int n) {
  int i = blockIdx.x * blockDim.x + threadIdx.x;
  if (i < n) dst[i] = __float2bfloat16(src[i]);
}

// Pack B matrix (KxN logical, value = transposed ? src[n*ld+k] : src[k*ld+n])
// into per-lane WMMA fragment layout: [(tn*Ktiles+tk)*32 + lane][16 bf16].
__global__ void pack_b_frag(const float* __restrict__ src,
                            __hip_bfloat16* __restrict__ dst,
                            int Ktiles, int Ntiles, int src_ld, int transposed) {
  int idx = blockIdx.x * blockDim.x + threadIdx.x;
  int total = Ntiles * Ktiles * 32;
  if (idx >= total) return;
  int lane = idx & 31;
  int tk = (idx >> 5) % Ktiles;
  int tn = (idx >> 5) / Ktiles;
  int n = tn * 16 + (lane & 15);
  int kb = tk * 32 + (lane >> 4) * 8;
  __hip_bfloat16* out = dst + (size_t)idx * 16;
#pragma unroll
  for (int j = 0; j < 16; ++j) {
    int k = kb + j + (j >= 8 ? 8 : 0);
    float v = transposed ? src[(size_t)n * src_ld + k] : src[(size_t)k * src_ld + n];
    out[j] = __float2bfloat16(v);
  }
}

// ---- phase 1: uin[t][b][r] = u @ w_in^T + bias  (M=16384,N=1024,K=128) ------
__global__ void __launch_bounds__(256) esn_input_gemm(
    const __hip_bfloat16* __restrict__ u_bf, const __hip_bfloat16* __restrict__ pk_win,
    const float* __restrict__ wbias, float* __restrict__ uin) {
  int lane = threadIdx.x & 31, wid = threadIdx.x >> 5;
  int mtile = blockIdx.x;
  int ntile = blockIdx.y * 8 + wid;
  v8f acc = {0.f, 0.f, 0.f, 0.f, 0.f, 0.f, 0.f, 0.f};
#pragma unroll
  for (int tk = 0; tk < 4; ++tk) {
    v16bf a = load_a_frag(u_bf + (size_t)mtile * 16 * 128 + tk * 32, 128, lane);
    v16bf b = load_b_frag(pk_win + ((size_t)(ntile * 4 + tk) * 32 + lane) * 16);
    acc = wmma_bf16(a, b, acc);
  }
  int n = ntile * 16 + (lane & 15);
  int hl = lane >> 4;
  float bias = wbias[n];
#pragma unroll
  for (int v = 0; v < 8; ++v) {
    int m = v + 8 * hl;
    int row = mtile * 16 + m;         // row = b*T + t (u is (B,T,I))
    int bidx = row >> 11;             // /T
    int t = row & 2047;               // %T
    uin[((size_t)t * 8 + bidx) * 1024 + n] = acc[v] + bias;
  }
}

// ---- phase 2: reservoir recurrence, 16 WGs x 4 waves, W slice in LDS --------
__global__ void __launch_bounds__(128) esn_reservoir(
    const __hip_bfloat16* __restrict__ pk_w, const float* __restrict__ uin,
    __hip_bfloat16* __restrict__ xbuf, __hip_bfloat16* __restrict__ res,
    unsigned int* __restrict__ bar) {
  extern __shared__ char smem_raw[];
  __hip_bfloat16* wlds = (__hip_bfloat16*)smem_raw;   // 4 ntiles * 32 ktiles * 512 = 128KB
  const int wg = blockIdx.x;                          // 0..15 -> owns n in [wg*64, wg*64+64)
  const int lane = threadIdx.x & 31;
  const int wid = threadIdx.x >> 5;                   // 0..3
  // Stage this WG's packed W slice (128KB) into LDS with one TDM descriptor.
  if (threadIdx.x < 32) {
    tdm_copy_to_lds(pk_w + (size_t)wg * 4 * 32 * 512, 0u, 4 * 32 * 512 * 2 / 8);
    __builtin_amdgcn_s_wait_tensorcnt(0);
  }
  __syncthreads();
  const int ntile = wg * 4 + wid;
  const int n = ntile * 16 + (lane & 15);
  const int hl = lane >> 4;
  for (int t = 0; t < 2048; ++t) {
    const __hip_bfloat16* xcur = xbuf + (size_t)(t & 1) * 16 * 1024;
    __hip_bfloat16* xnxt = xbuf + (size_t)((t + 1) & 1) * 16 * 1024;
    if (hl == 0 && t + 1 < 2048)  // pull next step's input drive toward L0/L2
      __builtin_prefetch(&uin[((size_t)(t + 1) * 8) * 1024 + n], 0, 1);
    v8f acc = {0.f, 0.f, 0.f, 0.f, 0.f, 0.f, 0.f, 0.f};
#pragma unroll 4
    for (int tk = 0; tk < 32; ++tk) {
      v16bf a = load_a_frag(xcur + tk * 32, 1024, lane);
      v16bf b = load_b_frag(wlds + ((size_t)(wid * 32 + tk) * 32 + lane) * 16);
      acc = wmma_bf16(a, b, acc);
    }
#pragma unroll
    for (int v = 0; v < 8; ++v) {
      int m = v + 8 * hl;                 // rows 8..15 are zero padding
      float pre = acc[v];
      if (hl == 0) pre += uin[((size_t)t * 8 + m) * 1024 + n];
      float xv = tanhf(pre);
      __hip_bfloat16 xb = __float2bfloat16(xv);
      xnxt[(size_t)m * 1024 + n] = xb;    // pad rows: tanh(0)=0 stays 0
      if (hl == 0) res[((size_t)t * 8 + m) * 1024 + n] = xb;
    }
    grid_sync(bar, 16u * (unsigned)(t + 1));
  }
}

// ---- phase 3: pca = res @ w_pca  (M=16384,N=256,K=1024) ---------------------
__global__ void __launch_bounds__(512) esn_pca(
    const __hip_bfloat16* __restrict__ res, const __hip_bfloat16* __restrict__ pk_wpca,
    float* __restrict__ pca_f, __hip_bfloat16* __restrict__ pca_bf) {
  int lane = threadIdx.x & 31, wid = threadIdx.x >> 5;  // 16 waves = 16 n-tiles
  int mtile = blockIdx.x;
  v8f acc = {0.f, 0.f, 0.f, 0.f, 0.f, 0.f, 0.f, 0.f};
#pragma unroll 4
  for (int tk = 0; tk < 32; ++tk) {
    v16bf a = load_a_frag(res + (size_t)mtile * 16 * 1024 + tk * 32, 1024, lane);
    v16bf b = load_b_frag(pk_wpca + ((size_t)(wid * 32 + tk) * 32 + lane) * 16);
    acc = wmma_bf16(a, b, acc);
  }
  int n = wid * 16 + (lane & 15);
  int hl = lane >> 4;
#pragma unroll
  for (int v = 0; v < 8; ++v) {
    int m = v + 8 * hl;
    int row = mtile * 16 + m;           // row = t*8 + b
    pca_f[(size_t)row * 256 + n] = acc[v];
    int t = row >> 3, b = row & 7;
    pca_bf[((size_t)t * 16 + b) * 256 + n] = __float2bfloat16(acc[v]);  // 16-row padded
  }
}

// ---- phase 4: gated recurrence, 2 WGs x 8 waves, Wzp/Wzh in LDS -------------
__global__ void __launch_bounds__(256) esn_gate(
    const __hip_bfloat16* __restrict__ pk_wzp, const __hip_bfloat16* __restrict__ pk_wzh,
    const float* __restrict__ bz, const float* __restrict__ pca_f,
    const __hip_bfloat16* __restrict__ pca_bf,
    __hip_bfloat16* __restrict__ hbf, float* __restrict__ hf,
    float* __restrict__ out, unsigned int* __restrict__ bar) {
  extern __shared__ char smem_raw[];
  __hip_bfloat16* zp = (__hip_bfloat16*)smem_raw;   // 8*8*512 = 64KB
  __hip_bfloat16* zh = zp + 8 * 8 * 512;            // 64KB
  int wg = blockIdx.x;                              // 0..1 -> owns n in [wg*128, +128)
  int lane = threadIdx.x & 31, wid = threadIdx.x >> 5;  // 0..7
  // Stage both gate matrices with two in-order TDM descriptors (one wave issues).
  if (threadIdx.x < 32) {
    tdm_copy_to_lds(pk_wzp + (size_t)wg * 8 * 8 * 512, 0u, 8 * 8 * 512 * 2 / 8);
    tdm_copy_to_lds(pk_wzh + (size_t)wg * 8 * 8 * 512, 8 * 8 * 512 * 2,
                    8 * 8 * 512 * 2 / 8);
    __builtin_amdgcn_s_wait_tensorcnt(0);
  }
  __syncthreads();
  int ntile = wg * 8 + wid;
  int n = ntile * 16 + (lane & 15);
  int hl = lane >> 4;
  float bzn = bz[n];
  for (int t = 0; t < 2048; ++t) {
    const __hip_bfloat16* hcur = hbf + (size_t)(t & 1) * 16 * 256;
    __hip_bfloat16* hnxt = hbf + (size_t)((t + 1) & 1) * 16 * 256;
    const float* hfc = hf + (size_t)(t & 1) * 16 * 256;
    float* hfn = hf + (size_t)((t + 1) & 1) * 16 * 256;
    const __hip_bfloat16* pt = pca_bf + (size_t)t * 16 * 256;
    v8f acc = {0.f, 0.f, 0.f, 0.f, 0.f, 0.f, 0.f, 0.f};
#pragma unroll
    for (int tk = 0; tk < 8; ++tk) {    // p @ Wzp^T
      v16bf a = load_a_frag(pt + tk * 32, 256, lane);
      v16bf b = load_b_frag(zp + ((size_t)(wid * 8 + tk) * 32 + lane) * 16);
      acc = wmma_bf16(a, b, acc);
    }
#pragma unroll
    for (int tk = 0; tk < 8; ++tk) {    // + h @ Wzh^T
      v16bf a = load_a_frag(hcur + tk * 32, 256, lane);
      v16bf b = load_b_frag(zh + ((size_t)(wid * 8 + tk) * 32 + lane) * 16);
      acc = wmma_bf16(a, b, acc);
    }
#pragma unroll
    for (int v = 0; v < 8; ++v) {
      int m = v + 8 * hl;
      float z = 1.0f / (1.0f + expf(-(acc[v] + bzn)));
      float p = (hl == 0) ? pca_f[((size_t)t * 8 + m) * 256 + n] : 0.0f;
      float hp = hfc[(size_t)m * 256 + n];
      float hn = (1.0f - z) * hp + z * p;    // pad rows: (1-z)*0 + z*0 = 0
      hfn[(size_t)m * 256 + n] = hn;
      hnxt[(size_t)m * 256 + n] = __float2bfloat16(hn);
      if (hl == 0) out[((size_t)m * 2048 + t) * 256 + n] = hn;  // (B,T,H)
    }
    grid_sync(bar, 2u * (unsigned)(t + 1));
  }
}

// ---- host ------------------------------------------------------------------
extern "C" void kernel_launch(void* const* d_in, const int* in_sizes, int n_in,
                              void* d_out, int out_size, void* d_ws, size_t ws_size,
                              hipStream_t stream) {
  (void)in_sizes; (void)n_in; (void)out_size; (void)ws_size;
  const float* u     = (const float*)d_in[0];
  const float* w_in  = (const float*)d_in[1];
  const float* w     = (const float*)d_in[2];
  const float* wbias = (const float*)d_in[3];
  const float* wpca  = (const float*)d_in[4];
  const float* wzp   = (const float*)d_in[5];
  const float* wzh   = (const float*)d_in[6];
  const float* bz    = (const float*)d_in[7];
  float* out = (float*)d_out;

  char* ws = (char*)d_ws;
  size_t off = 0;
  auto take = [&](size_t bytes) -> char* {
    char* p = ws + off;
    off += (bytes + 255) & ~(size_t)255;
    return p;
  };
  __hip_bfloat16* u_bf    = (__hip_bfloat16*)take(16384ull * 128 * 2);
  __hip_bfloat16* pk_win  = (__hip_bfloat16*)take(64ull * 4 * 512 * 2);
  __hip_bfloat16* pk_w    = (__hip_bfloat16*)take(64ull * 32 * 512 * 2);
  __hip_bfloat16* pk_wpca = (__hip_bfloat16*)take(16ull * 32 * 512 * 2);
  __hip_bfloat16* pk_wzp  = (__hip_bfloat16*)take(16ull * 8 * 512 * 2);
  __hip_bfloat16* pk_wzh  = (__hip_bfloat16*)take(16ull * 8 * 512 * 2);
  float* uin              = (float*)take(2048ull * 8 * 1024 * 4);
  __hip_bfloat16* xbuf    = (__hip_bfloat16*)take(2ull * 16 * 1024 * 2);
  __hip_bfloat16* resb    = (__hip_bfloat16*)take(2048ull * 8 * 1024 * 2);
  float* pca_f            = (float*)take(2048ull * 8 * 256 * 4);
  __hip_bfloat16* pca_bf  = (__hip_bfloat16*)take(2048ull * 16 * 256 * 2);
  __hip_bfloat16* hbf     = (__hip_bfloat16*)take(2ull * 16 * 256 * 2);
  float* hf               = (float*)take(2ull * 16 * 256 * 4);
  unsigned int* bars      = (unsigned int*)take(512);

  hipMemsetAsync(xbuf, 0, 2ull * 16 * 1024 * 2, stream);
  hipMemsetAsync(pca_bf, 0, 2048ull * 16 * 256 * 2, stream);
  hipMemsetAsync(hbf, 0, 2ull * 16 * 256 * 2, stream);
  hipMemsetAsync(hf, 0, 2ull * 16 * 256 * 4, stream);
  hipMemsetAsync(bars, 0, 512, stream);

  f32_to_bf16_k<<<(16384 * 128 + 255) / 256, 256, 0, stream>>>(u, u_bf, 16384 * 128);
  pack_b_frag<<<(64 * 4 * 32 + 255) / 256, 256, 0, stream>>>(w_in, pk_win, 4, 64, 128, 1);
  pack_b_frag<<<(64 * 32 * 32 + 255) / 256, 256, 0, stream>>>(w, pk_w, 32, 64, 1024, 1);
  pack_b_frag<<<(16 * 32 * 32 + 255) / 256, 256, 0, stream>>>(wpca, pk_wpca, 32, 16, 256, 0);
  pack_b_frag<<<(16 * 8 * 32 + 255) / 256, 256, 0, stream>>>(wzp, pk_wzp, 8, 16, 256, 1);
  pack_b_frag<<<(16 * 8 * 32 + 255) / 256, 256, 0, stream>>>(wzh, pk_wzh, 8, 16, 256, 1);

  esn_input_gemm<<<dim3(1024, 8), 256, 0, stream>>>(u_bf, pk_win, wbias, uin);
  esn_reservoir<<<16, 128, 131072, stream>>>(pk_w, uin, xbuf, resb, bars);
  esn_pca<<<1024, 512, 0, stream>>>(resb, pk_wpca, pca_f, pca_bf);
  esn_gate<<<2, 256, 131072, stream>>>(pk_wzp, pk_wzh, bz, pca_f, pca_bf, hbf, hf,
                                       out, bars + 64);
}
